// PairwiseTreeLSTMModel_37469294691121
// MI455X (gfx1250) — matrix operand order repaired and checked by reading, not compiled, not run on hardware
//
#include <hip/hip_runtime.h>
#include <hip/hip_bf16.h>

// ---------------------------------------------------------------------------
// Pairwise ChildSum TreeLSTM on MI455X (gfx1250), wave32 + v_wmma_f32_16x16x32_f16.
//
// Forest topology is deterministic (complete 4-ary trees, depth 5):
//   local parent(p) children = 4p+1..4p+4 ; depth-d block starts at (4^d-1)/3.
// Level l (leaves=0) == depth d=5-l, cnt/tree = 4^d (power of 4 -> shift/mask maps).
//
// GEMM operands f16 (fp32 accumulate), c-state fp32. Weights pre-packed into
// WMMA B-operand fragments: tile (kt,nt) -> lane L holds B[k=kt*32+(L>>4)*8+{0..7,16..23}][n=n0+(L&15)]
// as 16 contiguous halves (CDNA5 ISA 16-bit B layout). A loaded from
// row-major f16: lane L = row m0+(L&15), halves = K {hi*8+0..7, 16+hi*8+0..7}.
// C/D frag: reg v <-> row m0 + v + (L>=16)*8, col n0 + (L&15).
// ---------------------------------------------------------------------------

#define NTREES 128
#define TNODES 1365
#define NN (NTREES * TNODES)   // 174720
#define HD 128

typedef __attribute__((ext_vector_type(16))) _Float16 v16h;
typedef __attribute__((ext_vector_type(8)))  _Float16 v8h;
typedef __attribute__((ext_vector_type(4)))  _Float16 v4h;
typedef __attribute__((ext_vector_type(8)))  float    v8f;

__device__ __forceinline__ float sigm_f(float x) { return 1.0f / (1.0f + __expf(-x)); }
__device__ __forceinline__ float tanh_f(float x) {
  x = fminf(15.0f, fmaxf(-15.0f, x));
  float e = __expf(2.0f * x);
  return (e - 1.0f) / (e + 1.0f);
}

__device__ __forceinline__ v16h load_a_frag(const _Float16* __restrict Ar, int hi) {
  v8h lo = *(const v8h*)(Ar + hi * 8);
  v8h hh = *(const v8h*)(Ar + 16 + hi * 8);
  return __builtin_shufflevector(lo, hh, 0,1,2,3,4,5,6,7,8,9,10,11,12,13,14,15);
}

// ---- pack fp32 weights [nOut x kIn] into WMMA B fragments (f16) ------------
// frag[(kt*NT+nt)*512 + lane*16 + j] = W[nt*16+(lane&15)][kt*32+(lane>>4)*8+(j&7)+(j>>3)*16]
// kIn split: k<128 from P0, k>=128 from P1 (both row-stride 128).
__global__ void k_pack_wfrag(const float* __restrict P0, const float* __restrict P1,
                             _Float16* __restrict out, int KT, int NT) {
  int t = blockIdx.x * blockDim.x + threadIdx.x;
  if (t >= KT * NT * 32) return;
  int lane = t & 31, tile = t >> 5;
  int nt = tile % NT, kt = tile / NT;
  int k = kt * 32;
  const float* src = (k < 128) ? P0 : P1;
  int kk = ((k < 128) ? k : (k - 128)) + (lane >> 4) * 8;
  _Float16* o = out + (size_t)tile * 512 + lane * 16;
#pragma unroll
  for (int j = 0; j < 16; ++j) o[j] = (_Float16)src[(size_t)(nt * 16 + (lane & 15)) * 128 + kk + (j & 7) + (j >> 3) * 16];
}

// ---- embedding gather -> row-major f16 x[node][128] ------------------------
__global__ void k_embed(const int* __restrict tokens, const float* __restrict emb,
                        _Float16* __restrict xbuf) {
  int t = blockIdx.x * blockDim.x + threadIdx.x;
  if (t >= NN * 32) return;
  int node = t >> 5, g = t & 31;
  float4 v = *(const float4*)(emb + (size_t)tokens[node] * HD + g * 4);
  v4h o; o[0] = (_Float16)v.x; o[1] = (_Float16)v.y; o[2] = (_Float16)v.z; o[3] = (_Float16)v.w;
  *(v4h*)(xbuf + (size_t)node * HD + g * 4) = o;
}

// ---- per-edge forget gate + c_sum reduction --------------------------------
// Rows = child nodes (level l-1). f = sigmoid(h_child @ U_f^T + b_f);
// csum[parent_row][col] = sum over 4 consecutive child rows of f * c_child.
// Parent of child row r' is r' >> 2 (holds globally: cnt divisible by 4).
__global__ void k_forget(const _Float16* __restrict hbuf, const float* __restrict cbuf,
                         const _Float16* __restrict uffrag, const float* __restrict b_f,
                         float* __restrict csum, int Mtiles, int startC, int shiftC) {
  int tile = blockIdx.x * blockDim.y + threadIdx.y;
  if (tile >= Mtiles) return;
  int lane = threadIdx.x;
  int m0 = tile * 16, nt = blockIdx.y, n0 = nt * 16;
  int hi = lane >> 4;
  int maskC = (1 << shiftC) - 1;
  int rl = m0 + (lane & 15);
  const _Float16* Arow = hbuf + ((size_t)(rl >> shiftC) * TNODES + startC + (rl & maskC)) * HD;

  v8f acc = {0.f, 0.f, 0.f, 0.f, 0.f, 0.f, 0.f, 0.f};
#pragma unroll
  for (int kt = 0; kt < 4; ++kt) {
    v16h a = load_a_frag(Arow + kt * 32, hi);
    v16h b = *(const v16h*)(uffrag + (size_t)(kt * 8 + nt) * 512 + lane * 16);
    acc = __builtin_amdgcn_wmma_f32_16x16x32_f16(false, a, false, b, (short)0, acc, false, false);
  }
  int col = n0 + (lane & 15);
  float bias = b_f[col];
  float s0 = 0.f, s1 = 0.f;
#pragma unroll
  for (int v = 0; v < 8; ++v) {
    int rr = m0 + v + hi * 8;
    size_t node = (size_t)(rr >> shiftC) * TNODES + startC + (rr & maskC);
    float f = sigm_f(acc[v] + bias);
    float p = f * cbuf[node * HD + col];
    if (v < 4) s0 += p; else s1 += p;
  }
  int p0 = (m0 >> 2) + hi * 2;   // parent rows (each written by exactly one lane)
  csum[(size_t)p0 * HD + col] = s0;
  csum[(size_t)(p0 + 1) * HD + col] = s1;
}

// ---- h_tild[r] = sum of 4 child h rows (f16 -> fp32 -> f16) ----------------
__global__ void k_htild(const _Float16* __restrict hbuf, _Float16* __restrict htbuf,
                        int nRows, int startC, int shiftC) {
  int t = blockIdx.x * blockDim.x + threadIdx.x;
  if (t >= nRows * 32) return;
  int r = t >> 5, g = t & 31;
  int maskC = (1 << shiftC) - 1;
  float s0 = 0.f, s1 = 0.f, s2 = 0.f, s3 = 0.f;
#pragma unroll
  for (int k = 0; k < 4; ++k) {
    int rc = 4 * r + k;
    size_t node = (size_t)(rc >> shiftC) * TNODES + startC + (rc & maskC);
    v4h h = *(const v4h*)(hbuf + node * HD + g * 4);
    s0 += (float)h[0]; s1 += (float)h[1]; s2 += (float)h[2]; s3 += (float)h[3];
  }
  v4h o; o[0] = (_Float16)s0; o[1] = (_Float16)s1; o[2] = (_Float16)s2; o[3] = (_Float16)s3;
  *(v4h*)(htbuf + (size_t)r * HD + g * 4) = o;
}

// ---- fused iou GEMM + LSTM gate --------------------------------------------
// iou = [x | h_tild] @ [W_iou ; U_iou]^T + b_iou  (K = 256; leaves use K = 128)
// One wave owns a 16-row block and one 16-col chunk of each of i/o/u.
__global__ void k_gate(const _Float16* __restrict xbuf, const _Float16* __restrict htbuf,
                       const _Float16* __restrict w2frag, const float* __restrict b_iou,
                       const float* __restrict csum, float* __restrict cbuf,
                       _Float16* __restrict hbuf, int Mtiles, int startL, int shiftL, int KT) {
  int tile = blockIdx.x * blockDim.y + threadIdx.y;
  if (tile >= Mtiles) return;
  int lane = threadIdx.x;
  int m0 = tile * 16, cc = blockIdx.y;
  int hi = lane >> 4;
  int maskL = (1 << shiftL) - 1;
  int rl = m0 + (lane & 15);
  const _Float16* Xrow = xbuf + ((size_t)(rl >> shiftL) * TNODES + startL + (rl & maskL)) * HD;
  const _Float16* Hrow = htbuf + (size_t)rl * HD;

  v8f ai = {0.f,0.f,0.f,0.f,0.f,0.f,0.f,0.f};
  v8f ao = ai, au = ai;
#pragma unroll 4
  for (int kt = 0; kt < KT; ++kt) {
    const _Float16* Ar = (kt < 4) ? (Xrow + kt * 32) : (Hrow + (kt - 4) * 32);
    v16h a = load_a_frag(Ar, hi);
    const _Float16* wb = w2frag + (size_t)kt * 24 * 512 + (size_t)lane * 16;
    v16h bi = *(const v16h*)(wb + (size_t)cc * 512);
    v16h bo = *(const v16h*)(wb + (size_t)(8 + cc) * 512);
    v16h bu = *(const v16h*)(wb + (size_t)(16 + cc) * 512);
    ai = __builtin_amdgcn_wmma_f32_16x16x32_f16(false, a, false, bi, (short)0, ai, false, false);
    ao = __builtin_amdgcn_wmma_f32_16x16x32_f16(false, a, false, bo, (short)0, ao, false, false);
    au = __builtin_amdgcn_wmma_f32_16x16x32_f16(false, a, false, bu, (short)0, au, false, false);
  }
  int col = cc * 16 + (lane & 15);
  float Bi = b_iou[col], Bo = b_iou[128 + col], Bu = b_iou[256 + col];
#pragma unroll
  for (int v = 0; v < 8; ++v) {
    int rr = m0 + v + hi * 8;
    size_t node = (size_t)(rr >> shiftL) * TNODES + startL + (rr & maskL);
    float cs = (KT == 4) ? 0.f : csum[(size_t)rr * HD + col];
    float ig = sigm_f(ai[v] + Bi);
    float og = sigm_f(ao[v] + Bo);
    float ug = tanh_f(au[v] + Bu);
    float cn = ig * ug + cs;
    float hn = og * tanh_f(cn);
    cbuf[node * HD + col] = cn;
    hbuf[node * HD + col] = (_Float16)hn;
  }
}

// ---- per-tree mean readout -------------------------------------------------
__global__ void k_readout(const _Float16* __restrict hbuf, float* __restrict fout) {
  int tree = blockIdx.x, col = threadIdx.x;
  const _Float16* p = hbuf + (size_t)tree * TNODES * HD + col;
  float s = 0.f;
  for (int i = 0; i < TNODES; ++i) s += (float)p[(size_t)i * HD];
  fout[tree * HD + col] = s * (1.0f / (float)TNODES);
}

// ---- head: euc -> linear -> leaky_relu -> softmax --------------------------
__global__ void k_head(const float* __restrict f1, const float* __restrict f2,
                       const float* __restrict Wf, const float* __restrict bfv,
                       float* __restrict out) {
  __shared__ float e[HD];
  __shared__ float lg[2];
  int tree = blockIdx.x, t = threadIdx.x;
  float d = f1[tree * HD + t] - f2[tree * HD + t];
  e[t] = d * d;
  __syncthreads();
  if (t < 2) {
    float s = bfv[t];
    for (int k = 0; k < HD; ++k) s += e[k] * Wf[t * HD + k];
    lg[t] = (s > 0.f) ? s : 0.01f * s;
  }
  __syncthreads();
  if (t == 0) {
    float m = fmaxf(lg[0], lg[1]);
    float e0 = __expf(lg[0] - m), e1 = __expf(lg[1] - m);
    float inv = 1.f / (e0 + e1);
    out[tree * 2 + 0] = e0 * inv;
    out[tree * 2 + 1] = e1 * inv;
  }
}

extern "C" void kernel_launch(void* const* d_in, const int* in_sizes, int n_in,
                              void* d_out, int out_size, void* d_ws, size_t ws_size,
                              hipStream_t stream) {
  (void)in_sizes; (void)n_in; (void)out_size; (void)ws_size;

  const int*   tokens[2] = {(const int*)d_in[0], (const int*)d_in[1]};
  const float* emb       = (const float*)d_in[6];
  const float* W_iou[2]  = {(const float*)d_in[7],  (const float*)d_in[12]};
  const float* U_iou[2]  = {(const float*)d_in[8],  (const float*)d_in[13]};
  const float* b_iou[2]  = {(const float*)d_in[9],  (const float*)d_in[14]};
  const float* U_f[2]    = {(const float*)d_in[10], (const float*)d_in[15]};
  const float* b_f[2]    = {(const float*)d_in[11], (const float*)d_in[16]};
  const float* Wf        = (const float*)d_in[17];
  const float* bfv       = (const float*)d_in[18];
  float* out = (float*)d_out;

  // workspace carve-up (256B aligned)
  char* w = (char*)d_ws;
  size_t off = 0;
  auto take = [&](size_t bytes) { void* p = w + off; off = (off + bytes + 255) & ~(size_t)255; return p; };
  _Float16* w2frag[2] = {(_Float16*)take(8 * 24 * 512 * 2), (_Float16*)take(8 * 24 * 512 * 2)};
  _Float16* uffrag[2] = {(_Float16*)take(4 * 8 * 512 * 2),  (_Float16*)take(4 * 8 * 512 * 2)};
  _Float16* xbuf  = (_Float16*)take((size_t)NN * HD * 2);
  _Float16* hbuf  = (_Float16*)take((size_t)NN * HD * 2);
  float*    cbuf  = (float*)   take((size_t)NN * HD * 4);
  _Float16* htbuf = (_Float16*)take((size_t)32768 * HD * 2);
  float*    csbuf = (float*)   take((size_t)32768 * HD * 4);
  float*    fbuf  = (float*)   take((size_t)2 * NTREES * HD * 4);

  static const int S5[6] = {0, 1, 5, 21, 85, 341};   // depth -> local start

  for (int m = 0; m < 2; ++m) {
    k_pack_wfrag<<<(8 * 24 * 32 + 255) / 256, 256, 0, stream>>>(W_iou[m], U_iou[m], w2frag[m], 8, 24);
    k_pack_wfrag<<<(4 * 8 * 32 + 255) / 256, 256, 0, stream>>>(U_f[m], U_f[m], uffrag[m], 4, 8);
  }

  for (int m = 0; m < 2; ++m) {
    k_embed<<<(NN * 32 + 255) / 256, 256, 0, stream>>>(tokens[m], emb, xbuf);
    for (int l = 0; l <= 5; ++l) {
      int d = 5 - l;                        // depth of level-l nodes
      int startL = S5[d], shiftL = 2 * d;
      int nRows  = NTREES << shiftL;
      int Mtiles = nRows / 16;
      if (l > 0) {
        int dc = d + 1;                     // children depth
        int startC = S5[dc], shiftC = 2 * dc;
        int Mc = (NTREES << shiftC) / 16;
        k_forget<<<dim3((Mc + 3) / 4, 8), dim3(32, 4), 0, stream>>>(
            hbuf, cbuf, uffrag[m], b_f[m], csbuf, Mc, startC, shiftC);
        k_htild<<<(nRows * 32 + 255) / 256, 256, 0, stream>>>(hbuf, htbuf, nRows, startC, shiftC);
      }
      k_gate<<<dim3((Mtiles + 3) / 4, 8), dim3(32, 4), 0, stream>>>(
          xbuf, htbuf, w2frag[m], b_iou[m], csbuf, cbuf, hbuf,
          Mtiles, startL, shiftL, (l == 0) ? 4 : 8);
    }
    k_readout<<<NTREES, HD, 0, stream>>>(hbuf, fbuf + (size_t)m * NTREES * HD);
  }

  k_head<<<NTREES, HD, 0, stream>>>(fbuf, fbuf + (size_t)NTREES * HD, Wf, bfv, out);
}
